// SPVUnet_49478023250391
// MI455X (gfx1250) — matrix operand order repaired and checked by reading, not compiled, not run on hardware
//
#include <hip/hip_runtime.h>
#include <math.h>

// ---------------------------------------------------------------------------
// MI455X / gfx1250 implementation of the SPVUnet reference.
//  * All conv3d / strided-conv / transposed-conv / 1x1 / point-MLP layers run
//    through one implicit-GEMM kernel using v_wmma_f32_16x16x32_f16 (wave32).
//  * Activations & weights f16 (channels-last); accumulation f32.
//  * Macro tile: 128(M) x 64(N), 4 waves/WG; each wave computes 2 M-frags x
//    4 N-tiles = 8 WMMAs per 32-wide K step. LDS-staged tiles; fragment loads
//    follow the CDNA5 ISA 16x32 A / 32x16 B VGPR layouts (2x b128 each).
//  * Mode is a template parameter; all grid-decode math is shift/mask (grids
//    are powers of two), k->(tap,ci) decode hoisted to once per k-step.
//  * conv_transpose(k=2,s=2) == per-output-phase 1x1 GEMM (gridDim.z = 8).
//  * scatter-mean via global f32 atomics; trilinear gather elementwise.
// Input flattening assumption: setup_inputs() dict insertion order
// (coords, feats, t, params), params flattened as a JAX pytree with keys
// sorted alphabetically at every nesting level -> 118 leaves.
// ---------------------------------------------------------------------------

typedef _Float16 h8   __attribute__((ext_vector_type(8)));
typedef _Float16 v16h __attribute__((ext_vector_type(16)));
typedef float    v8f  __attribute__((ext_vector_type(8)));

#define MODE_POINT 0   // A [M,Ci] dense GEMM (1x1 conv / point MLP)
#define MODE_CONV3 1   // 3x3x3 SAME conv, implicit im2col, K=27*Ci
#define MODE_DOWN2 2   // 2x2x2 stride-2 VALID conv, K=8*Ci
#define MODE_UPT2  3   // 2x2x2 stride-2 transposed conv, per-phase 1x1

// ------------------------------ small kernels ------------------------------

__global__ void cvt_f32_f16(const float* __restrict__ in, _Float16* __restrict__ out, int n) {
  int i = blockIdx.x * blockDim.x + threadIdx.x;
  if (i < n) out[i] = (_Float16)in[i];
}

__global__ void fill0(float* __restrict__ p, int n) {
  int i = blockIdx.x * blockDim.x + threadIdx.x;
  if (i < n) p[i] = 0.f;
}

// sinusoidal t-embedding (TEMB=32) + 2-layer MLP (bn+relu between) -> emb[B,128]
__global__ void time_emb_kernel(const int* __restrict__ t,
                                const float* __restrict__ w1, const float* __restrict__ b1,
                                const float* __restrict__ bng, const float* __restrict__ bnb,
                                const float* __restrict__ w2, const float* __restrict__ b2,
                                float* __restrict__ emb) {
  __shared__ float te[32];
  __shared__ float h1[128];
  const int b = blockIdx.x, j = threadIdx.x;
  if (j < 16) {
    float f = __expf(-9.2103403719761836f * (float)j / 16.0f);
    float a = (float)t[b] * f;
    te[j] = __cosf(a);
    te[j + 16] = __sinf(a);
  }
  __syncthreads();
  float s = b1[j];
  for (int i = 0; i < 32; ++i) s += te[i] * w1[i * 128 + j];
  s = s * bng[j] + bnb[j];
  h1[j] = s > 0.f ? s : 0.f;
  __syncthreads();
  float o = b2[j];
  for (int i = 0; i < 128; ++i) o += h1[i] * w2[i * 128 + j];
  emb[b * 128 + j] = o;
}

// abias[b][j] = sum_k emb[b][k]*tw[k][j] + tb[j]
__global__ void emb_proj_kernel(const float* __restrict__ emb, const float* __restrict__ tw,
                                const float* __restrict__ tb, float* __restrict__ ab,
                                int ni, int total) {
  int idx = blockIdx.x * blockDim.x + threadIdx.x;
  if (idx >= total) return;
  int b = idx / ni, j = idx - b * ni;
  float s = tb[j];
  const float* e = emb + b * 128;
  for (int k = 0; k < 128; ++k) s += e[k] * tw[k * ni + j];
  ab[idx] = s;
}

// scatter-add point features into grid sums (f32 atomics)
__global__ void p2v_scatter(const float* __restrict__ coords,
                            const float* __restrict__ f32s, const _Float16* __restrict__ f16s,
                            float* __restrict__ sum, int Bn, int Np, int r, int C, float scale) {
  int idx = blockIdx.x * blockDim.x + threadIdx.x;
  int total = Bn * Np * C;
  if (idx >= total) return;
  int c = idx % C, p = (idx / C) % Np, b = idx / (C * Np);
  const float* pc = coords + ((size_t)b * Np + p) * 3;
  int i0 = min(max((int)floorf(pc[0] * scale), 0), r - 1);
  int i1 = min(max((int)floorf(pc[1] * scale), 0), r - 1);
  int i2 = min(max((int)floorf(pc[2] * scale), 0), r - 1);
  size_t voff = (((size_t)b * r + i0) * r + i1) * r + i2;
  float v = f32s ? f32s[(size_t)idx] : (float)f16s[((size_t)b * Np + p) * C + c];
  atomicAdd(&sum[voff * C + c], v);
}

__global__ void p2v_count(const float* __restrict__ coords, float* __restrict__ cnt,
                          int Bn, int Np, int r, float scale) {
  int idx = blockIdx.x * blockDim.x + threadIdx.x;
  if (idx >= Bn * Np) return;
  int p = idx % Np, b = idx / Np;
  const float* pc = coords + ((size_t)b * Np + p) * 3;
  int i0 = min(max((int)floorf(pc[0] * scale), 0), r - 1);
  int i1 = min(max((int)floorf(pc[1] * scale), 0), r - 1);
  int i2 = min(max((int)floorf(pc[2] * scale), 0), r - 1);
  atomicAdd(&cnt[(((size_t)b * r + i0) * r + i1) * r + i2], 1.0f);
}

__global__ void p2v_norm(const float* __restrict__ sum, const float* __restrict__ cnt,
                         _Float16* __restrict__ outg, int C, int total) {
  int idx = blockIdx.x * blockDim.x + threadIdx.x;
  if (idx >= total) return;
  outg[idx] = (_Float16)(sum[idx] / fmaxf(cnt[idx / C], 1.0f));
}

// trilinear gather: grid [B,r,r,r,C] f16 -> out [B*Np, C] f16
__global__ void v2p_kernel(const _Float16* __restrict__ g, const float* __restrict__ coords,
                           _Float16* __restrict__ outp, int Bn, int Np, int r, int C, float scale) {
  int idx = blockIdx.x * blockDim.x + threadIdx.x;
  int total = Bn * Np * C;
  if (idx >= total) return;
  int c = idx % C, p = (idx / C) % Np, b = idx / (C * Np);
  const float* pc = coords + ((size_t)b * Np + p) * 3;
  float lim = (float)r - 1.0001f;
  float p0 = fminf(fmaxf(pc[0] * scale, 0.f), lim);
  float p1 = fminf(fmaxf(pc[1] * scale, 0.f), lim);
  float p2 = fminf(fmaxf(pc[2] * scale, 0.f), lim);
  int a0 = (int)floorf(p0), a1 = (int)floorf(p1), a2 = (int)floorf(p2);
  float f0 = p0 - a0, f1 = p1 - a1, f2 = p2 - a2;
  float acc = 0.f;
  for (int d0 = 0; d0 < 2; ++d0)
    for (int d1 = 0; d1 < 2; ++d1)
      for (int d2 = 0; d2 < 2; ++d2) {
        int i0 = min(a0 + d0, r - 1), i1 = min(a1 + d1, r - 1), i2 = min(a2 + d2, r - 1);
        float w = (d0 ? f0 : 1.f - f0) * (d1 ? f1 : 1.f - f1) * (d2 ? f2 : 1.f - f2);
        acc += w * (float)g[((((size_t)b * r + i0) * r + i1) * r + i2) * C + c];
      }
  outp[((size_t)b * Np + p) * C + c] = (_Float16)acc;
}

__global__ void concat2(const _Float16* __restrict__ a, const _Float16* __restrict__ b,
                        _Float16* __restrict__ o, int M, int C1, int C2) {
  int idx = blockIdx.x * blockDim.x + threadIdx.x;
  int Ct = C1 + C2;
  if (idx >= M * Ct) return;
  int m = idx / Ct, c = idx - m * Ct;
  o[idx] = (c < C1) ? a[(size_t)m * C1 + c] : b[(size_t)m * C2 + (c - C1)];
}

// ---------------------- WMMA implicit-GEMM conv kernel ----------------------
// 128 threads = 4 waves. Macro tile 128(M) x 64(N); wave w owns M rows
// [32w, 32w+32) as two 16-row fragments and all four 16-wide N tiles:
// 8 x v_wmma_f32_16x16x32_f16 per 32-wide K step.

template <int MODE>
__global__ __launch_bounds__(128)
void conv_gemm_wmma(const _Float16* __restrict__ A, const _Float16* __restrict__ Wt,
                    const float* __restrict__ g, const float* __restrict__ bb,
                    const float* __restrict__ lb, const _Float16* __restrict__ res,
                    const float* __restrict__ abias,
                    _Float16* __restrict__ o16, float* __restrict__ o32,
                    int dlog, int ciLog, int Ci, int Co,
                    int Mtot, int Ktot, int relu, int resPost) {
  __shared__ __align__(16) _Float16 As[128][40];  // [m][k], 80B rows (16B aligned)
  __shared__ __align__(16) _Float16 Bs[64][40];   // [n][k] (transposed weight tile)
  __shared__ int4 Rw[128];                        // per-row voxel decode (b,z,y,x)

  const int tid = threadIdx.x;
  const int wave = tid >> 5, lane = tid & 31;
  const int m0 = blockIdx.x * 128;
  const int n0 = blockIdx.y * 64;
  const int phase = blockIdx.z;                   // UPT2 only
  const int D = 1 << dlog, maskD = D - 1;
  const _Float16* Wp = Wt + ((MODE == MODE_UPT2) ? (size_t)phase * Ci * Co : 0);

  // one-time per-row voxel decode (shift/mask only; grids are pow2)
  if constexpr (MODE == MODE_CONV3) {
    int m = m0 + tid;
    int b = m >> (3 * dlog);
    int rr = m & ((1 << (3 * dlog)) - 1);
    Rw[tid] = make_int4(b, rr >> (2 * dlog), (rr >> dlog) & maskD, rr & maskD);
  } else if constexpr (MODE == MODE_DOWN2) {
    int dh = dlog - 1;                            // output grid is D/2
    int m = m0 + tid;
    int b = m >> (3 * dh);
    int rr = m & ((1 << (3 * dh)) - 1);
    int mh = (1 << dh) - 1;
    Rw[tid] = make_int4(b, (rr >> (2 * dh)) << 1, ((rr >> dh) & mh) << 1, (rr & mh) << 1);
  }
  __syncthreads();

  v8f acc[2][4];
  for (int a = 0; a < 2; ++a)
    for (int i = 0; i < 4; ++i)
      for (int e = 0; e < 8; ++e) acc[a][i][e] = 0.f;

  const int klA = tid & 31;        // this thread's fixed k-lane within the step
  const int mlbA = tid >> 5;       // row base; rows mlbA + 4j
  const int nB = tid & 63;         // B staging: fixed n column
  const int ngB = n0 + nB;

  for (int kk = 0; kk < Ktot; kk += 32) {
    const int k = kk + klA;
    const bool kvalid = (k < Ktot);
    // k -> (tap, ci) decode: once per k-step per thread (shift/mask or /3)
    int tap = 0, ci = k;
    if constexpr (MODE == MODE_CONV3 || MODE == MODE_DOWN2) {
      if (ciLog >= 0) { tap = k >> ciLog; ci = k & (Ci - 1); }
      else            { tap = k / 3;      ci = k - tap * 3; }   // stem: Ci==3
    }
    int dz = 0, dy = 0, dx = 0;
    if constexpr (MODE == MODE_CONV3) {
      dz = tap / 9 - 1; dy = (tap / 3) % 3 - 1; dx = tap % 3 - 1;
    } else if constexpr (MODE == MODE_DOWN2) {
      dz = tap >> 2; dy = (tap >> 1) & 1; dx = tap & 1;
    }
    // ---- stage A tile (128 x 32) ----
#pragma unroll 4
    for (int j = 0; j < 32; ++j) {
      int ml = mlbA + 4 * j;
      int m = m0 + ml;
      float v = 0.f;
      if (m < Mtot && kvalid) {
        if constexpr (MODE == MODE_POINT || MODE == MODE_UPT2) {
          v = (float)A[(size_t)m * Ci + k];
        } else if constexpr (MODE == MODE_CONV3) {
          int4 ri = Rw[ml];
          int zz = ri.y + dz, yy = ri.z + dy, xx = ri.w + dx;
          if ((unsigned)zz < (unsigned)D && (unsigned)yy < (unsigned)D &&
              (unsigned)xx < (unsigned)D) {
            v = (float)A[((size_t)((ri.x * D + zz) * D + yy) * D + xx) * Ci + ci];
            if (abias) v += abias[ri.x * Ci + ci];   // fused time-embedding bias
          }
        } else {  // MODE_DOWN2 (always in bounds)
          int4 ri = Rw[ml];
          int zz = ri.y + dz, yy = ri.z + dy, xx = ri.w + dx;
          v = (float)A[((size_t)((ri.x * D + zz) * D + yy) * D + xx) * Ci + ci];
        }
      }
      As[ml][klA] = (_Float16)v;
    }
    // ---- stage B tile transposed: Bs[n][k] from Wt[k][n] ----
#pragma unroll 4
    for (int j = 0; j < 16; ++j) {
      int kr = (tid >> 6) + 2 * j;
      int kg = kk + kr;
      _Float16 v = (_Float16)0.f;
      if (kg < Ktot && ngB < Co) v = Wp[(size_t)kg * Co + ngB];
      Bs[nB][kr] = v;
    }
    __syncthreads();
    if (kk + 32 < Ktot)  // global_prefetch next weight tile
      __builtin_prefetch(&Wp[(size_t)(kk + 32) * Co + n0], 0, 1);

    // ---- fragment loads per CDNA5 ISA layouts + 8x WMMA ----
    union Frag { v16h v; h8 h[2]; };
    Frag a0, a1;
    {
      // A 16x32 f16: lane<16 -> row=lane, K {0..7,16..23}; lane>=16 -> K {8..15,24..31}
      int row = wave * 32 + (lane & 15);
      int kb = (lane < 16) ? 0 : 8;
      a0.h[0] = *reinterpret_cast<const h8*>(&As[row][kb]);
      a0.h[1] = *reinterpret_cast<const h8*>(&As[row][kb + 16]);
      a1.h[0] = *reinterpret_cast<const h8*>(&As[row + 16][kb]);
      a1.h[1] = *reinterpret_cast<const h8*>(&As[row + 16][kb + 16]);
    }
#pragma unroll
    for (int nt = 0; nt < 4; ++nt) {
      // B 32x16 f16: lane<16 -> col=lane, K {0..7,16..23}; lane>=16 -> K {8..15,24..31}
      Frag bf;
      int col = nt * 16 + (lane & 15);
      int kb = (lane < 16) ? 0 : 8;
      bf.h[0] = *reinterpret_cast<const h8*>(&Bs[col][kb]);
      bf.h[1] = *reinterpret_cast<const h8*>(&Bs[col][kb + 16]);
      acc[0][nt] = __builtin_amdgcn_wmma_f32_16x16x32_f16(false, a0.v, false, bf.v,
                                                          (short)0, acc[0][nt], false, false);
      acc[1][nt] = __builtin_amdgcn_wmma_f32_16x16x32_f16(false, a1.v, false, bf.v,
                                                          (short)0, acc[1][nt], false, false);
    }
    __syncthreads();
  }

  // ---- epilogue: D layout = VGPR r: lanes<16 M=r, lanes>=16 M=8+r; N=lane&15 ----
  const int noff = lane & 15;
#pragma unroll
  for (int at = 0; at < 2; ++at) {
    const int mbase = m0 + wave * 32 + at * 16 + ((lane < 16) ? 0 : 8);
#pragma unroll
    for (int r = 0; r < 8; ++r) {
      int m = mbase + r;
      if (m >= Mtot) continue;
      size_t obase;
      if constexpr (MODE == MODE_UPT2) {
        int b = m >> (3 * dlog);
        int rr = m & ((1 << (3 * dlog)) - 1);
        int z = rr >> (2 * dlog), y = (rr >> dlog) & maskD, x = rr & maskD;
        int Do = D << 1;
        int oz = 2 * z + (phase >> 2), oy = 2 * y + ((phase >> 1) & 1), ox = 2 * x + (phase & 1);
        obase = ((size_t)((b * Do + oz) * Do + oy) * Do + ox) * Co;
      } else {
        obase = (size_t)m * Co;
      }
#pragma unroll
      for (int nt = 0; nt < 4; ++nt) {
        int n = n0 + nt * 16 + noff;
        if (n >= Co) continue;
        float v = acc[at][nt][r];
        if (lb) v += lb[n];
        if (res && !resPost) v += (float)res[(size_t)m * Co + n];   // pre-BN residual
        if (g) v = v * g[n] + bb[n];                                // BN affine
        if (res && resPost) v += (float)res[(size_t)m * Co + n];    // post-BN residual
        if (relu) v = v > 0.f ? v : 0.f;
        if (o16) o16[obase + n] = (_Float16)v;
        else     o32[obase + n] = v;
      }
    }
  }
}

// ------------------------------- host driver -------------------------------

extern "C" void kernel_launch(void* const* d_in, const int* in_sizes, int n_in,
                              void* d_out, int out_size, void* d_ws, size_t ws_size,
                              hipStream_t stream) {
  if (n_in < 118) return;
  const int Bn = 4, Np = 2048;
  const float* coords = (const float*)d_in[0];
  const float* feats  = (const float*)d_in[1];
  const int*   tsteps = (const int*)d_in[2];

  // pytree leaf bases (alphabetical at every level; see header comment)
  const int D1 = 3, D2B = 11, D3B = 19, D4B = 27, EMB = 35, OUTB = 41, OUTU = 46,
            PM1 = 49, PM2 = 53, PM3 = 57, PM4 = 61, STEM = 65,
            U0 = 74, U1 = 85, U2 = 96, U3 = 107;
  // down: +0 bn.b +1 bn.g +2 bnd.b +3 bnd.g +4 tb +5 tw +6 w +7 wd
  // up:   +0 bn.b +1 bn.g +2 bnf.b +3 bnf.g +4 bnu.b +5 bnu.g +6 tb +7 tw +8 w +9 wf +10 wu
  // emb:  +0 b1 +1 b2 +2 bn1.b +3 bn1.g +4 w1 +5 w2
  // out:  +0 b2 +1 bn.b +2 bn.g +3 w1 +4 w2 ; out_unet: +0 bn.b +1 bn.g +2 w
  // pm:   +0 b +1 bn.b +2 bn.g +3 w
  // stem: +0 bn1.b +1 bn1.g +2 bn2.b +3 bn2.g +4 bnid.b +5 bnid.g +6 w1 +7 w2 +8 wid
  auto F = [&](int i) { return (const float*)d_in[i]; };

  char* base = (char*)d_ws;
  size_t off = 0;
  auto alloc = [&](size_t bytes) {
    void* p = base + off;
    off = (off + bytes + 255) & ~(size_t)255;
    return p;
  };
  auto a16 = [&](size_t n) { return (_Float16*)alloc(n * 2); };
  auto a32 = [&](size_t n) { return (float*)alloc(n * 4); };

  const size_t V64 = (size_t)Bn * 64 * 64 * 64;
  const size_t V32 = (size_t)Bn * 32 * 32 * 32;
  const size_t V16v = (size_t)Bn * 16 * 16 * 16;
  const size_t V8v = (size_t)Bn * 8 * 8 * 8;
  const size_t V4v = (size_t)Bn * 4 * 4 * 4;
  const size_t P = (size_t)Bn * Np;

  // arenas + persistent skip tensors (f16, channels-last)
  _Float16* A0 = a16(V64 * 32);
  _Float16* A1 = a16(V64 * 64);
  _Float16* A2 = a16(V64 * 32);
  _Float16* hx1 = a16(V64 * 32);
  _Float16* hx2 = a16(V32 * 64);
  _Float16* hx3p = a16(V16v * 128);
  _Float16* hx4 = a16(V8v * 256);
  _Float16* z1 = a16(P * 32);
  _Float16* z2 = a16(P * 128);
  _Float16* z3 = a16(P * 512);
  _Float16* z4 = a16(P * 128);
  _Float16* z5 = a16(P * 32);
  _Float16* zt = a16(P * 512);
  _Float16* zh = a16(P * 32);
  float* sumA = a32(V64 * 32);
  float* cntA = a32(V64);
  float* embf = a32(Bn * 128);
  float* abd1 = a32(Bn * 32);
  float* abd2 = a32(Bn * 64);
  float* abd3 = a32(Bn * 128);
  float* abd4 = a32(Bn * 256);
  float* abu0 = a32(Bn * 512);
  float* abu1 = a32(Bn * 256);
  float* abu2 = a32(Bn * 128);
  float* abu3 = a32(Bn * 64);

  auto cvt = [&](int leaf) {
    size_t n = (size_t)in_sizes[leaf];
    _Float16* d = a16(n);
    cvt_f32_f16<<<(unsigned)((n + 255) / 256), 256, 0, stream>>>(F(leaf), d, (int)n);
    return (const _Float16*)d;
  };
  // weight conversions (f32 -> f16, contiguous [Ktot, Co] layout already)
  const _Float16 *w_st1 = cvt(STEM + 6), *w_st2 = cvt(STEM + 7), *w_stid = cvt(STEM + 8);
  const _Float16 *w_d1 = cvt(D1 + 6), *w_d1d = cvt(D1 + 7);
  const _Float16 *w_d2 = cvt(D2B + 6), *w_d2d = cvt(D2B + 7);
  const _Float16 *w_d3 = cvt(D3B + 6), *w_d3d = cvt(D3B + 7);
  const _Float16 *w_d4 = cvt(D4B + 6), *w_d4d = cvt(D4B + 7);
  const _Float16 *w_u0 = cvt(U0 + 8), *w_u0f = cvt(U0 + 9), *w_u0u = cvt(U0 + 10);
  const _Float16 *w_u1 = cvt(U1 + 8), *w_u1f = cvt(U1 + 9), *w_u1u = cvt(U1 + 10);
  const _Float16 *w_u2 = cvt(U2 + 8), *w_u2f = cvt(U2 + 9), *w_u2u = cvt(U2 + 10);
  const _Float16 *w_u3 = cvt(U3 + 8), *w_u3f = cvt(U3 + 9), *w_u3u = cvt(U3 + 10);
  const _Float16 *w_pm1 = cvt(PM1 + 3), *w_pm2 = cvt(PM2 + 3);
  const _Float16 *w_pm3 = cvt(PM3 + 3), *w_pm4 = cvt(PM4 + 3);
  const _Float16 *w_ou = cvt(OUTU + 2);
  const _Float16 *w_o1 = cvt(OUTB + 3), *w_o2 = cvt(OUTB + 4);

  // time embedding + per-block embedding projections
  time_emb_kernel<<<Bn, 128, 0, stream>>>(tsteps, F(EMB + 4), F(EMB + 0), F(EMB + 3),
                                          F(EMB + 2), F(EMB + 5), F(EMB + 1), embf);
  auto eproj = [&](int twl, int tbl, float* ab, int ni) {
    int tot = Bn * ni;
    emb_proj_kernel<<<(tot + 127) / 128, 128, 0, stream>>>(embf, F(twl), F(tbl), ab, ni, tot);
  };
  eproj(D1 + 5, D1 + 4, abd1, 32);
  eproj(D2B + 5, D2B + 4, abd2, 64);
  eproj(D3B + 5, D3B + 4, abd3, 128);
  eproj(D4B + 5, D4B + 4, abd4, 256);
  eproj(U0 + 7, U0 + 6, abu0, 512);
  eproj(U1 + 7, U1 + 6, abu1, 256);
  eproj(U2 + 7, U2 + 6, abu2, 128);
  eproj(U3 + 7, U3 + 6, abu3, 64);

  auto zero = [&](float* p, size_t n) {
    fill0<<<(unsigned)((n + 255) / 256), 256, 0, stream>>>(p, (int)n);
  };
  auto p2v = [&](const float* f32s, const _Float16* f16s, int r, int C, float scale,
                 _Float16* outg) {
    size_t r3 = (size_t)Bn * r * r * r;
    zero(sumA, r3 * C);
    zero(cntA, r3);
    int tot = (int)(P * C);
    p2v_scatter<<<(tot + 255) / 256, 256, 0, stream>>>(coords, f32s, f16s, sumA, Bn, Np, r, C, scale);
    p2v_count<<<((int)P + 255) / 256, 256, 0, stream>>>(coords, cntA, Bn, Np, r, scale);
    int tn = (int)(r3 * C);
    p2v_norm<<<(tn + 255) / 256, 256, 0, stream>>>(sumA, cntA, outg, C, tn);
  };
  auto v2p = [&](const _Float16* grid, int r, int C, float scale, _Float16* outp) {
    int tot = (int)(P * C);
    v2p_kernel<<<(tot + 255) / 256, 256, 0, stream>>>(grid, coords, outp, Bn, Np, r, C, scale);
  };
  auto cat = [&](const _Float16* a, int C1, const _Float16* b, int C2, size_t M, _Float16* o) {
    int tot = (int)(M * (C1 + C2));
    concat2<<<(tot + 255) / 256, 256, 0, stream>>>(a, b, o, (int)M, C1, C2);
  };
  auto gemm = [&](int mode, const _Float16* Aact, const _Float16* Wt, int Din, int Ci, int Co,
                  int Mpoint, const float* g, const float* bbv, const float* lb,
                  const _Float16* res, int resPost, const float* ab,
                  _Float16* o16, float* o32, int relu) {
    int Mtot, Ktot, phases = 1;
    if (mode == MODE_POINT) { Mtot = Mpoint; Ktot = Ci; }
    else if (mode == MODE_CONV3) { Mtot = Bn * Din * Din * Din; Ktot = 27 * Ci; }
    else if (mode == MODE_DOWN2) { int dh = Din / 2; Mtot = Bn * dh * dh * dh; Ktot = 8 * Ci; }
    else { Mtot = Bn * Din * Din * Din; Ktot = Ci; phases = 8; }
    int dlog = (Din > 1) ? __builtin_ctz(Din) : 0;
    int ciLog = (Ci & (Ci - 1)) ? -1 : __builtin_ctz(Ci);
    dim3 grid((Mtot + 127) / 128, (Co + 63) / 64, phases);
    dim3 blk(128);
    if (mode == MODE_POINT)
      conv_gemm_wmma<MODE_POINT><<<grid, blk, 0, stream>>>(Aact, Wt, g, bbv, lb, res, ab, o16, o32,
          dlog, ciLog, Ci, Co, Mtot, Ktot, relu, resPost);
    else if (mode == MODE_CONV3)
      conv_gemm_wmma<MODE_CONV3><<<grid, blk, 0, stream>>>(Aact, Wt, g, bbv, lb, res, ab, o16, o32,
          dlog, ciLog, Ci, Co, Mtot, Ktot, relu, resPost);
    else if (mode == MODE_DOWN2)
      conv_gemm_wmma<MODE_DOWN2><<<grid, blk, 0, stream>>>(Aact, Wt, g, bbv, lb, res, ab, o16, o32,
          dlog, ciLog, Ci, Co, Mtot, Ktot, relu, resPost);
    else
      conv_gemm_wmma<MODE_UPT2><<<grid, blk, 0, stream>>>(Aact, Wt, g, bbv, lb, res, ab, o16, o32,
          dlog, ciLog, Ci, Co, Mtot, Ktot, relu, resPost);
  };

  // ---------------------------- forward graph ----------------------------
  p2v(feats, nullptr, 64, 3, 64.f, A0);                                           // x0
  gemm(MODE_POINT, A0, w_stid, 1, 3, 32, (int)V64, F(STEM + 5), F(STEM + 4),
       nullptr, nullptr, 0, nullptr, A1, nullptr, 0);                             // ident
  gemm(MODE_CONV3, A0, w_st1, 64, 3, 32, 0, F(STEM + 1), F(STEM + 0),
       nullptr, nullptr, 0, nullptr, A2, nullptr, 1);                             // stem conv1
  gemm(MODE_CONV3, A2, w_st2, 64, 32, 32, 0, F(STEM + 3), F(STEM + 2),
       nullptr, A1, 1, nullptr, hx1, nullptr, 1);                                 // x1 (post-BN res)
  v2p(hx1, 64, 32, 64.f, z1);                                                     // z1
  gemm(MODE_CONV3, hx1, w_d1, 64, 32, 32, 0, F(D1 + 1), F(D1 + 0),
       nullptr, nullptr, 0, abd1, A0, nullptr, 1);                                // d1 conv (+temb)
  gemm(MODE_DOWN2, A0, w_d1d, 64, 32, 64, 0, F(D1 + 3), F(D1 + 2),
       nullptr, nullptr, 0, nullptr, hx2, nullptr, 1);                            // x2
  gemm(MODE_CONV3, hx2, w_d2, 32, 64, 64, 0, F(D2B + 1), F(D2B + 0),
       nullptr, nullptr, 0, abd2, A0, nullptr, 1);
  gemm(MODE_DOWN2, A0, w_d2d, 32, 64, 128, 0, F(D2B + 3), F(D2B + 2),
       nullptr, nullptr, 0, nullptr, A1, nullptr, 1);                             // x3
  v2p(A1, 16, 128, 16.f, zt);
  gemm(MODE_POINT, z1, w_pm1, 1, 32, 128, (int)P, F(PM1 + 2), F(PM1 + 1),
       F(PM1 + 0), zt, 0, nullptr, z2, nullptr, 1);                               // z2
  p2v(nullptr, z2, 16, 128, 16.f, hx3p);                                          // x3p
  gemm(MODE_CONV3, hx3p, w_d3, 16, 128, 128, 0, F(D3B + 1), F(D3B + 0),
       nullptr, nullptr, 0, abd3, A0, nullptr, 1);
  gemm(MODE_DOWN2, A0, w_d3d, 16, 128, 256, 0, F(D3B + 3), F(D3B + 2),
       nullptr, nullptr, 0, nullptr, hx4, nullptr, 1);                            // x4
  gemm(MODE_CONV3, hx4, w_d4, 8, 256, 256, 0, F(D4B + 1), F(D4B + 0),
       nullptr, nullptr, 0, abd4, A0, nullptr, 1);
  gemm(MODE_DOWN2, A0, w_d4d, 8, 256, 512, 0, F(D4B + 3), F(D4B + 2),
       nullptr, nullptr, 0, nullptr, A1, nullptr, 1);                             // x4n (4^3,512)
  v2p(A1, 4, 512, 4.f, zt);
  gemm(MODE_POINT, z2, w_pm2, 1, 128, 512, (int)P, F(PM2 + 2), F(PM2 + 1),
       F(PM2 + 0), zt, 0, nullptr, z3, nullptr, 1);                               // z3
  p2v(nullptr, z3, 4, 512, 4.f, A2);                                              // x4p
  cat(A2, 512, A1, 512, V4v, A0);                                                 // concat
  gemm(MODE_POINT, A0, w_u0f, 1, 1024, 512, (int)V4v, F(U0 + 3), F(U0 + 2),
       nullptr, nullptr, 0, nullptr, A1, nullptr, 1);                             // u0 wf
  gemm(MODE_CONV3, A1, w_u0, 4, 512, 512, 0, F(U0 + 1), F(U0 + 0),
       nullptr, nullptr, 0, abu0, A2, nullptr, 1);
  gemm(MODE_UPT2, A2, w_u0u, 4, 512, 256, 0, F(U0 + 5), F(U0 + 4),
       nullptr, nullptr, 0, nullptr, A0, nullptr, 1);                             // 8^3,256
  cat(hx4, 256, A0, 256, V8v, A1);
  gemm(MODE_POINT, A1, w_u1f, 1, 512, 256, (int)V8v, F(U1 + 3), F(U1 + 2),
       nullptr, nullptr, 0, nullptr, A2, nullptr, 1);
  gemm(MODE_CONV3, A2, w_u1, 8, 256, 256, 0, F(U1 + 1), F(U1 + 0),
       nullptr, nullptr, 0, abu1, A0, nullptr, 1);
  gemm(MODE_UPT2, A0, w_u1u, 8, 256, 128, 0, F(U1 + 5), F(U1 + 4),
       nullptr, nullptr, 0, nullptr, A1, nullptr, 1);                             // x5 16^3,128
  v2p(A1, 16, 128, 16.f, zt);
  gemm(MODE_POINT, z3, w_pm3, 1, 512, 128, (int)P, F(PM3 + 2), F(PM3 + 1),
       F(PM3 + 0), zt, 0, nullptr, z4, nullptr, 1);                               // z4
  p2v(nullptr, z4, 16, 128, 16.f, A2);                                            // x5p
  cat(A2, 128, hx3p, 128, V16v, A0);
  gemm(MODE_POINT, A0, w_u2f, 1, 256, 128, (int)V16v, F(U2 + 3), F(U2 + 2),
       nullptr, nullptr, 0, nullptr, A1, nullptr, 1);
  gemm(MODE_CONV3, A1, w_u2, 16, 128, 128, 0, F(U2 + 1), F(U2 + 0),
       nullptr, nullptr, 0, abu2, A2, nullptr, 1);
  gemm(MODE_UPT2, A2, w_u2u, 16, 128, 64, 0, F(U2 + 5), F(U2 + 4),
       nullptr, nullptr, 0, nullptr, A0, nullptr, 1);                             // x6 32^3,64
  cat(A0, 64, hx2, 64, V32, A1);
  gemm(MODE_POINT, A1, w_u3f, 1, 128, 64, (int)V32, F(U3 + 3), F(U3 + 2),
       nullptr, nullptr, 0, nullptr, A2, nullptr, 1);
  gemm(MODE_CONV3, A2, w_u3, 32, 64, 64, 0, F(U3 + 1), F(U3 + 0),
       nullptr, nullptr, 0, abu3, A0, nullptr, 1);
  gemm(MODE_UPT2, A0, w_u3u, 32, 64, 32, 0, F(U3 + 5), F(U3 + 4),
       nullptr, nullptr, 0, nullptr, A2, nullptr, 1);                             // x7 64^3,32
  v2p(A2, 64, 32, 64.f, zt);
  gemm(MODE_POINT, z4, w_pm4, 1, 128, 32, (int)P, F(PM4 + 2), F(PM4 + 1),
       F(PM4 + 0), zt, 0, nullptr, z5, nullptr, 1);                               // z5
  p2v(nullptr, z5, 64, 32, 64.f, A0);                                             // x8p
  cat(A0, 32, hx1, 32, V64, A1);
  gemm(MODE_CONV3, A1, w_ou, 64, 64, 32, 0, F(OUTU + 1), F(OUTU + 0),
       nullptr, nullptr, 0, nullptr, A0, nullptr, 1);                             // x8
  v2p(A0, 64, 32, 64.f, zt);                                                      // z6
  gemm(MODE_POINT, zt, w_o1, 1, 32, 32, (int)P, F(OUTB + 2), F(OUTB + 1),
       nullptr, nullptr, 0, nullptr, zh, nullptr, 1);                             // head hidden
  gemm(MODE_POINT, zh, w_o2, 1, 32, 3, (int)P, nullptr, nullptr,
       F(OUTB + 0), nullptr, 0, nullptr, nullptr, (float*)d_out, 0);              // [B,N,3] f32
}
